// DeepSpeedMoE_87600152969311
// MI455X (gfx1250) — compile-verified, matrix operand order checked by reading, not run on hardware
//
#include <hip/hip_runtime.h>
#include <hip/hip_bf16.h>
#include <math.h>

// ---------------- problem constants (from reference) ----------------
#define S_TOK 16384   // B*N = 8*2048
#define DMODEL 1024
#define HFF   4096
#define NEXP  8
#define CAPC  2560    // int(S*1.25/E)
#define LOSS_COEF 0.01f

typedef __attribute__((ext_vector_type(16))) __bf16 v16bf;
typedef __attribute__((ext_vector_type(8)))  float  v8f;
typedef __attribute__((ext_vector_type(4)))  __bf16 bf4;

union V16 { uint4 u[2]; v16bf v; };

__device__ __forceinline__ float gelu_tanh(float x) {
    float x3 = x * x * x;
    float t = tanhf(0.7978845608028654f * (x + 0.044715f * x3));
    return 0.5f * x * (1.0f + t);
}

// ---------------- 1. zero small accumulators ----------------
__global__ void moe_init(float* me_sum, float* ce_sum) {
    int t = threadIdx.x;
    if (t < NEXP) { me_sum[t] = 0.0f; ce_sum[t] = 0.0f; }
}

// ---------------- 2. zero expert input buffers (bf16) ----------------
__global__ void moe_zero_buf(uint4* buf, long n16) {
    long i = (long)blockIdx.x * blockDim.x + threadIdx.x;
    if (i < n16) buf[i] = make_uint4(0u, 0u, 0u, 0u);
}

// ---------------- 3. transpose+convert f32 [e][R][Cc] -> bf16 [e][Cc][R] ----
__global__ void moe_transpose_cvt(const float* __restrict__ in,
                                  __bf16* __restrict__ out, int R, int Cc) {
    __shared__ float tile[32][33];
    int e  = blockIdx.z;
    int c0 = blockIdx.x * 32;
    int r0 = blockIdx.y * 32;
    const float* inp  = in  + (size_t)e * R * Cc;
    __bf16*      outp = out + (size_t)e * R * Cc;
    int tx = threadIdx.x, ty = threadIdx.y;   // block (32,8)
#pragma unroll
    for (int i = 0; i < 4; i++)
        tile[ty + 8 * i][tx] = inp[(size_t)(r0 + ty + 8 * i) * Cc + c0 + tx];
    __syncthreads();
#pragma unroll
    for (int i = 0; i < 4; i++)
        outp[(size_t)(c0 + ty + 8 * i) * R + r0 + tx] = (__bf16)tile[tx][ty + 8 * i];
}

// ---------------- 4. gating: softmax(x@Wg), top-2, loss stats ----------------
// one wave (32 lanes) per token, 8 waves per block
__global__ void moe_gating(const float* __restrict__ x,
                           const float* __restrict__ Wg,
                           int* __restrict__ idx1, int* __restrict__ idx2,
                           float* __restrict__ gate1, float* __restrict__ gate2,
                           float* me_sum, float* ce_sum) {
    __shared__ float wg[DMODEL * NEXP];           // 32 KB
    int t = threadIdx.x;
#pragma unroll
    for (int i = 0; i < (DMODEL * NEXP) / 256; i++)
        wg[t + i * 256] = Wg[t + i * 256];
    __syncthreads();

    int wave = t >> 5, lane = t & 31;
    int s = blockIdx.x * 8 + wave;
    const float* xr = x + (size_t)s * DMODEL;

    float acc[NEXP];
#pragma unroll
    for (int e = 0; e < NEXP; e++) acc[e] = 0.0f;

    for (int i = 0; i < DMODEL / 32; i++) {
        int d = lane + i * 32;
        float xv = xr[d];
#pragma unroll
        for (int e = 0; e < NEXP; e++) acc[e] += xv * wg[d * NEXP + e];
    }
#pragma unroll
    for (int off = 16; off > 0; off >>= 1)
#pragma unroll
        for (int e = 0; e < NEXP; e++) acc[e] += __shfl_xor(acc[e], off, 32);

    if (lane == 0) {
        float mx = acc[0];
#pragma unroll
        for (int e = 1; e < NEXP; e++) mx = fmaxf(mx, acc[e]);
        float g[NEXP], sum = 0.0f;
#pragma unroll
        for (int e = 0; e < NEXP; e++) { g[e] = expf(acc[e] - mx); sum += g[e]; }
        float inv = 1.0f / sum;
#pragma unroll
        for (int e = 0; e < NEXP; e++) g[e] *= inv;

        int i1 = 0; float v1 = g[0];
#pragma unroll
        for (int e = 1; e < NEXP; e++) if (g[e] > v1) { v1 = g[e]; i1 = e; }
        int i2 = -1; float v2 = -1.0f;
#pragma unroll
        for (int e = 0; e < NEXP; e++)
            if (e != i1 && g[e] > v2) { v2 = g[e]; i2 = e; }

        idx1[s] = i1; idx2[s] = i2;
        gate1[s] = v1; gate2[s] = v2;
#pragma unroll
        for (int e = 0; e < NEXP; e++) atomicAdd(&me_sum[e], g[e]);
        atomicAdd(&ce_sum[i1], 1.0f);
    }
}

// ---------------- 5. scan1: in-order top-1 positions + counts ----------------
__global__ void moe_scan1(const int* __restrict__ idx1, int* __restrict__ pos1,
                          int* __restrict__ count1) {
    __shared__ int sh[256];
    __shared__ int base[NEXP];
    int t = threadIdx.x;
    if (t < NEXP) base[t] = 0;
    __syncthreads();
    for (int ch = 0; ch < S_TOK / 256; ch++) {
        int s = ch * 256 + t;
        int e = idx1[s];
        sh[t] = e;
        __syncthreads();
        int p = base[e];
        for (int j = 0; j < t; j++) p += (sh[j] == e);
        pos1[s] = p;
        __syncthreads();
        if (t < NEXP) {
            int c = 0;
            for (int j = 0; j < 256; j++) c += (sh[j] == t);
            base[t] += c;
        }
        __syncthreads();
    }
    if (t < NEXP) count1[t] = base[t];
}

// ---------- 6. scan2: top-2 positions (offset by used1), finalize gates -------
__global__ void moe_scan2(const int* __restrict__ idx2,
                          const float* __restrict__ gate1,
                          const float* __restrict__ gate2,
                          const int* __restrict__ count1,
                          int* __restrict__ p1, int* __restrict__ p2,
                          float* __restrict__ g1, float* __restrict__ g2) {
    __shared__ int sh[256];
    __shared__ int base[NEXP];
    int t = threadIdx.x;
    if (t < NEXP) {
        int c = count1[t];
        base[t] = c < CAPC ? c : CAPC;   // used1[e]
    }
    __syncthreads();
    for (int ch = 0; ch < S_TOK / 256; ch++) {
        int s = ch * 256 + t;
        int e2 = idx2[s];
        sh[t] = e2;
        __syncthreads();
        int p2v = base[e2];
        for (int j = 0; j < t; j++) p2v += (sh[j] == e2);
        int p1v = p1[s];                  // raw pos1 written by scan1
        float k1 = (p1v < CAPC) ? 1.0f : 0.0f;
        float k2 = (p2v < CAPC) ? 1.0f : 0.0f;
        float gg1 = gate1[s], gg2 = gate2[s];
        float denom = gg1 * k1 + gg2 * k2 + 1e-9f;
        g1[s] = gg1 * k1 / denom;
        g2[s] = gg2 * k2 / denom;
        p1[s] = (p1v < CAPC) ? p1v : 0;
        p2[s] = (p2v < CAPC) ? p2v : 0;
        __syncthreads();
        if (t < NEXP) {
            int c = 0;
            for (int j = 0; j < 256; j++) c += (sh[j] == t);
            base[t] += c;
        }
        __syncthreads();
    }
}

// ---------------- 7. aux loss scalar ----------------
__global__ void moe_loss(const float* me_sum, const float* ce_sum, float* out_loss) {
    if (threadIdx.x == 0) {
        const float invS = 1.0f / (float)S_TOK;
        float acc = 0.0f;
#pragma unroll
        for (int e = 0; e < NEXP; e++)
            acc += (me_sum[e] * invS) * (ce_sum[e] * invS);
        out_loss[0] = LOSS_COEF * (acc / (float)NEXP) * (float)(NEXP * NEXP);
    }
}

// ---------------- 8. dispatch: scatter tokens to bf16 expert buffers ----------
__global__ void moe_dispatch(const float* __restrict__ x,
                             const int* __restrict__ idx1, const int* __restrict__ idx2,
                             const int* __restrict__ p1, const int* __restrict__ p2,
                             const float* __restrict__ g1, const float* __restrict__ g2,
                             __bf16* __restrict__ buf) {
    int s = blockIdx.x;
    int t = threadIdx.x;                       // 256 threads, 4 floats each
    const float4 v = ((const float4*)(x + (size_t)s * DMODEL))[t];
    bf4 b;
    b.x = (__bf16)v.x; b.y = (__bf16)v.y; b.z = (__bf16)v.z; b.w = (__bf16)v.w;
    if (g1[s] > 0.0f) {
        size_t off = ((size_t)idx1[s] * CAPC + p1[s]) * DMODEL + t * 4;
        *(bf4*)(buf + off) = b;
    }
    if (g2[s] > 0.0f) {
        size_t off = ((size_t)idx2[s] * CAPC + p2[s]) * DMODEL + t * 4;
        *(bf4*)(buf + off) = b;
    }
}

// ---------------- 9. grouped bf16 WMMA GEMM ----------------
// A   : bf16 [e][M][K] row-major (K contiguous)
// Bt  : bf16 [e][N][K] row-major (K contiguous) == B transposed
// out : GELU? bf16 [e][M][N] : f32 [e][M][N]
// block tile 128x128, BK=32; 8 waves (4 along M x 2 along N); wave = 2x4 WMMA tiles
template <bool GELU>
__global__ __launch_bounds__(256) void moe_gemm(const __bf16* __restrict__ A,
                                                const __bf16* __restrict__ Bt,
                                                const float* __restrict__ bias,
                                                void* __restrict__ Out,
                                                int M, int N, int K) {
    __shared__ __bf16 lA[128 * 40];   // padded row: 40 elems = 80 B (conflict-free b128)
    __shared__ __bf16 lB[128 * 40];

    const int e  = blockIdx.z;
    const int bm = blockIdx.y;
    const int bn = blockIdx.x;
    const int t  = threadIdx.x;
    const int lane = t & 31;
    const int w    = t >> 5;
    const int wm   = w & 3;       // 0..3 : 32 rows each
    const int wn   = w >> 2;      // 0..1 : 64 cols each
    const int lrow  = lane & 15;
    const int khalf = (lane >> 4) << 3;     // 0 or 8 (ISA 16-bit A/B layout)

    const __bf16* Ae = A  + (size_t)e * M * K + (size_t)(bm * 128) * K;
    const __bf16* Be = Bt + (size_t)e * N * K + (size_t)(bn * 128) * K;

    v8f acc[2][4] = {};

    for (int kt = 0; kt < K; kt += 32) {
        // stage 128x32 A and B tiles: 2 x uint4 per thread per matrix
#pragma unroll
        for (int i = 0; i < 2; i++) {
            int idx = t + i * 256;          // 0..511
            int row = idx >> 2;             // 0..127
            int c8  = (idx & 3) << 3;       // 0,8,16,24
            *(uint4*)&lA[row * 40 + c8] = *(const uint4*)&Ae[(size_t)row * K + kt + c8];
            *(uint4*)&lB[row * 40 + c8] = *(const uint4*)&Be[(size_t)row * K + kt + c8];
        }
        __syncthreads();

        V16 af[2], Bf[4];
#pragma unroll
        for (int f = 0; f < 2; f++) {
            int base = (wm * 32 + f * 16 + lrow) * 40 + khalf;
            af[f].u[0] = *(const uint4*)&lA[base];        // K 0-7  / 8-15
            af[f].u[1] = *(const uint4*)&lA[base + 16];   // K 16-23/ 24-31
        }
#pragma unroll
        for (int j = 0; j < 4; j++) {
            int base = (wn * 64 + j * 16 + lrow) * 40 + khalf;
            Bf[j].u[0] = *(const uint4*)&lB[base];
            Bf[j].u[1] = *(const uint4*)&lB[base + 16];
        }
#pragma unroll
        for (int f = 0; f < 2; f++)
#pragma unroll
            for (int j = 0; j < 4; j++)
                acc[f][j] = __builtin_amdgcn_wmma_f32_16x16x32_bf16(
                    false, af[f].v, false, Bf[j].v, (short)0, acc[f][j], false, false);
        __syncthreads();
    }

    // epilogue: C/D layout — row = v + 8*(lane>>4), col = lane&15
#pragma unroll
    for (int f = 0; f < 2; f++) {
#pragma unroll
        for (int j = 0; j < 4; j++) {
            int mg0 = bm * 128 + wm * 32 + f * 16 + ((lane >> 4) << 3);
            int ng  = bn * 128 + wn * 64 + j * 16 + lrow;
            float bv = bias[(size_t)e * N + ng];
#pragma unroll
            for (int v = 0; v < 8; v++) {
                float val = acc[f][j][v] + bv;
                size_t off = ((size_t)e * M + (mg0 + v)) * N + ng;
                if (GELU)
                    ((__bf16*)Out)[off] = (__bf16)gelu_tanh(val);
                else
                    ((float*)Out)[off] = val;
            }
        }
    }
}

// ---------------- 10. combine: y = g1*eo[i1,p1] + g2*eo[i2,p2] ----------------
__global__ void moe_combine(const float* __restrict__ eo,
                            const int* __restrict__ idx1, const int* __restrict__ idx2,
                            const int* __restrict__ p1, const int* __restrict__ p2,
                            const float* __restrict__ g1, const float* __restrict__ g2,
                            float* __restrict__ y) {
    int s = blockIdx.x;
    int t = threadIdx.x;                 // 256 threads, float4 each
    float w1 = g1[s], w2 = g2[s];
    const float4* r1 = (const float4*)(eo + ((size_t)idx1[s] * CAPC + p1[s]) * DMODEL);
    const float4* r2 = (const float4*)(eo + ((size_t)idx2[s] * CAPC + p2[s]) * DMODEL);
    float4 a = r1[t], b = r2[t], o;
    o.x = w1 * a.x + w2 * b.x;
    o.y = w1 * a.y + w2 * b.y;
    o.z = w1 * a.z + w2 * b.z;
    o.w = w1 * a.w + w2 * b.w;
    ((float4*)(y + (size_t)s * DMODEL))[t] = o;
}

// ---------------- launch ----------------
extern "C" void kernel_launch(void* const* d_in, const int* in_sizes, int n_in,
                              void* d_out, int out_size, void* d_ws, size_t ws_size,
                              hipStream_t stream) {
    (void)in_sizes; (void)n_in; (void)out_size; (void)ws_size;
    const float* x  = (const float*)d_in[0];   // [S, D]
    const float* Wg = (const float*)d_in[1];   // [D, E]
    const float* W1 = (const float*)d_in[2];   // [E, D, H]
    const float* b1 = (const float*)d_in[3];   // [E, H]
    const float* W2 = (const float*)d_in[4];   // [E, H, D]
    const float* b2 = (const float*)d_in[5];   // [E, D]
    float* y = (float*)d_out;                  // [S, D] + [1] loss

    char* ws = (char*)d_ws;
    // small arrays
    int*   idx1   = (int*)  (ws + 0);
    int*   idx2   = (int*)  (ws + 65536);
    float* gate1  = (float*)(ws + 131072);
    float* gate2  = (float*)(ws + 196608);
    int*   p1     = (int*)  (ws + 262144);
    int*   p2     = (int*)  (ws + 327680);
    float* g1     = (float*)(ws + 393216);
    float* g2     = (float*)(ws + 458752);
    int*   count1 = (int*)  (ws + 524288);
    float* me_sum = (float*)(ws + 524544);
    float* ce_sum = (float*)(ws + 524800);
    // big arrays (1 MB-aligned start)
    size_t off = 1048576;
    __bf16* W1t = (__bf16*)(ws + off); off += (size_t)NEXP * DMODEL * HFF * 2;   // [e][H][D]
    __bf16* W2t = (__bf16*)(ws + off); off += (size_t)NEXP * DMODEL * HFF * 2;   // [e][D][H]
    __bf16* buf = (__bf16*)(ws + off); off += (size_t)NEXP * CAPC * DMODEL * 2;  // [e][C][D]
    __bf16* h   = (__bf16*)(ws + off); off += (size_t)NEXP * CAPC * HFF * 2;     // [e][C][H]
    float*  eo  = (float*) (ws + off);                                           // [e][C][D]

    // 1. zero loss accumulators
    moe_init<<<1, 32, 0, stream>>>(me_sum, ce_sum);

    // 2. zero expert buffers
    {
        long n16 = (long)NEXP * CAPC * DMODEL * 2 / 16;
        moe_zero_buf<<<(unsigned)((n16 + 255) / 256), 256, 0, stream>>>((uint4*)buf, n16);
    }

    // 3. weight convert+transpose: W1 [e][D][H] -> W1t [e][H][D]; W2 -> W2t
    moe_transpose_cvt<<<dim3(HFF / 32, DMODEL / 32, NEXP), dim3(32, 8), 0, stream>>>(
        W1, W1t, DMODEL, HFF);
    moe_transpose_cvt<<<dim3(DMODEL / 32, HFF / 32, NEXP), dim3(32, 8), 0, stream>>>(
        W2, W2t, HFF, DMODEL);

    // 4. gating
    moe_gating<<<S_TOK / 8, 256, 0, stream>>>(x, Wg, idx1, idx2, gate1, gate2,
                                              me_sum, ce_sum);

    // 5-6. capacity scans
    moe_scan1<<<1, 256, 0, stream>>>(idx1, p1, count1);
    moe_scan2<<<1, 256, 0, stream>>>(idx2, gate1, gate2, count1, p1, p2, g1, g2);

    // 7. aux loss
    moe_loss<<<1, 32, 0, stream>>>(me_sum, ce_sum, y + (size_t)S_TOK * DMODEL);

    // 8. dispatch
    moe_dispatch<<<S_TOK, 256, 0, stream>>>(x, idx1, idx2, p1, p2, g1, g2, buf);

    // 9. expert FFN: h = gelu(buf@W1 + b1); eo = h@W2 + b2
    moe_gemm<true><<<dim3(HFF / 128, CAPC / 128, NEXP), 256, 0, stream>>>(
        buf, W1t, b1, (void*)h, CAPC, HFF, DMODEL);
    moe_gemm<false><<<dim3(DMODEL / 128, CAPC / 128, NEXP), 256, 0, stream>>>(
        h, W2t, b2, (void*)eo, CAPC, DMODEL, HFF);

    // 10. combine
    moe_combine<<<S_TOK, 256, 0, stream>>>(eo, idx1, idx2, p1, p2, g1, g2, y);
}
